// Attention_23467701305648
// MI455X (gfx1250) — matrix operand order repaired and checked by reading, not compiled, not run on hardware
//
#include <hip/hip_runtime.h>

typedef __attribute__((ext_vector_type(16))) __bf16 v16bf;
typedef __attribute__((ext_vector_type(8)))  float  v8f;

static __device__ __forceinline__ v8f wmma_bf16(v16bf a, v16bf b, v8f c) {
  // D(16x16,f32) = A(16x32,bf16) * B(32x16,bf16) + C
  return __builtin_amdgcn_wmma_f32_16x16x32_bf16(false, a, false, b, (short)0, c,
                                                 false, false);
}

#define BATCH    4
#define DIM      256
#define SEQ      2048
#define HEADS    8
#define DHEAD    64
#define HIDDEN   512
#define QKV_ROWS 1536
#define ATTN_SCALE 0.125f  // 64^-0.5

// ---------------------------------------------------------------------------
// TDM staging: DMA a (tile_d1 rows x tile_d0 elems) tile, row stride stride0,
// from global into LDS (packed row-major). dsz_code: 0=1B,1=2B,2=4B.
// ---------------------------------------------------------------------------
#if __has_builtin(__builtin_amdgcn_tensor_load_to_lds)
#define HAVE_TDM 1
typedef __attribute__((ext_vector_type(4))) unsigned int v4u;
typedef __attribute__((ext_vector_type(4))) int v4i;
typedef __attribute__((ext_vector_type(8))) int v8i;

static __device__ __forceinline__ void tdm_load_tile(const void* gsrc,
                                                     unsigned lds_off,
                                                     int dsz_code, int tile_d0,
                                                     int tile_d1, int tensor_d0,
                                                     int tensor_d1, int stride0) {
  unsigned long long ga = (unsigned long long)(size_t)gsrc;
  v4u g0;
  g0[0] = 1u;             // count=1, user-mode descriptor
  g0[1] = lds_off;        // lds_addr (bytes)
  g0[2] = (unsigned)ga;   // global_addr[31:0]
  g0[3] = ((unsigned)(ga >> 32) & 0x01FFFFFFu) | 0x80000000u;  // addr[56:32]|type=2
  v8i g1;
  g1[0] = dsz_code << 16;  // workgroup_mask=0, data_size
  g1[1] = (int)((tensor_d0 & 0xFFFF) << 16);
  g1[2] = (int)(((unsigned)tensor_d0 >> 16) | ((tensor_d1 & 0xFFFF) << 16));
  g1[3] = (int)(((unsigned)tensor_d1 >> 16) | ((tile_d0 & 0xFFFF) << 16));
  g1[4] = tile_d1 & 0xFFFF;  // tile_dim1 | tile_dim2=0
  g1[5] = stride0;           // tensor_dim0_stride[31:0]
  g1[6] = 0;
  g1[7] = 0;
  v4i gz = {0, 0, 0, 0};
#if defined(__clang_major__) && __clang_major__ >= 23
  v8i gz8 = {0, 0, 0, 0, 0, 0, 0, 0};
  __builtin_amdgcn_tensor_load_to_lds(g0, g1, gz, gz, gz8, 0);
#else
  __builtin_amdgcn_tensor_load_to_lds(g0, g1, gz, gz, 0);
#endif
}
#else
#define HAVE_TDM 0
#endif

// ---------------------------------------------------------------------------
// Kernel 1: qkv[b,o,n] = sum_c w_qkv[o,c] * x[b,c,n]   (f32 in, bf16 out)
// Block owns one 16-wide n-slice; x panel [256 x 16] f32 staged once via TDM.
// Each wave accumulates 12 row-tiles; B fragment reused 12x per k-step.
// ---------------------------------------------------------------------------
__global__ __launch_bounds__(256) void k_qkv_proj(const float* __restrict__ x,
                                                  const float* __restrict__ w,
                                                  __bf16* __restrict__ qkv) {
  __shared__ __attribute__((aligned(16))) float xs[DIM * 16];  // 16 KB

  const int lane = threadIdx.x & 31;
  const int wave = threadIdx.x >> 5;
  const int hi = lane >> 4;
  const int lx = lane & 15;

  const int bidx = blockIdx.x >> 7;  // / 128
  const int tn = blockIdx.x & 127;
  const float* xg = x + (size_t)bidx * DIM * SEQ + tn * 16;

#if HAVE_TDM
  if (wave == 0) {
    tdm_load_tile(xg, (unsigned)(size_t)(void*)xs, 2, 16, DIM, SEQ, DIM, SEQ);
    __builtin_amdgcn_s_wait_tensorcnt(0);
  }
#else
  {
    const float4* s4 = (const float4*)(xg + (size_t)threadIdx.x * SEQ);
    float4* dst = (float4*)&xs[threadIdx.x * 16];
    dst[0] = s4[0]; dst[1] = s4[1]; dst[2] = s4[2]; dst[3] = s4[3];
  }
#endif
  __syncthreads();

  const int tm0 = wave * 12;  // 96 row-tiles / 8 waves
  v8f acc[12] = {};

  for (int k0 = 0; k0 < DIM; k0 += 32) {  // dynamic loop: small static body
    v16bf bf;
#pragma unroll
    for (int e = 0; e < 16; ++e)
      bf[e] = (__bf16)xs[(k0 + hi * 8 + (e & 7) + (e >> 3) * 16) * 16 + lx];
#pragma unroll
    for (int t = 0; t < 12; ++t) {
      const float* wrow = w + (size_t)((tm0 + t) * 16 + lx) * DIM + k0;
      v16bf af;
#pragma unroll
      for (int e = 0; e < 8; ++e) {
        af[e]     = (__bf16)wrow[hi * 8 + e];
        af[e + 8] = (__bf16)wrow[16 + hi * 8 + e];
      }
      acc[t] = wmma_bf16(af, bf, acc[t]);
    }
  }

  __bf16* qb = qkv + (size_t)bidx * QKV_ROWS * SEQ + tn * 16 + lx;
#pragma unroll
  for (int t = 0; t < 12; ++t)
#pragma unroll
    for (int r = 0; r < 8; ++r)
      qb[(size_t)((tm0 + t) * 16 + hi * 8 + r) * SEQ] = (__bf16)acc[t][r];
}

// ---------------------------------------------------------------------------
// Kernel 2: fused flash attention. Block = 8 waves sharing (b,h); each wave
// owns one 16-wide i-tile. K/V j-tiles staged block-wide into LDS by the TDM
// (double-buffered, overlapped with WMMA compute). S^T(j,i) = K^T Q via WMMA;
// online softmax over j (xor-16 lane reduce); O(d,i) += V * P via WMMA.
// ---------------------------------------------------------------------------
__global__ __launch_bounds__(256) void k_attention(const __bf16* __restrict__ qkv,
                                                   __bf16* __restrict__ ao) {
  __shared__ __attribute__((aligned(16))) __bf16 sk[2][64 * 32];
  __shared__ __attribute__((aligned(16))) __bf16 sv[2][64 * 32];

  const int lane = threadIdx.x & 31;
  const int wave = threadIdx.x >> 5;
  const int hi = lane >> 4;
  const int lx = lane & 15;

  int wid = blockIdx.x * 8 + wave;  // 8 consecutive waves: same (b,h)
  int it = wid % 128;
  int h = (wid / 128) % HEADS;
  int bidx = wid / (128 * HEADS);

  const __bf16* qh = qkv + ((size_t)bidx * QKV_ROWS + (size_t)h * DHEAD) * SEQ;
  const __bf16* kh = qh + (size_t)HIDDEN * SEQ;
  const __bf16* vh = kh + (size_t)HIDDEN * SEQ;
  const int col = it * 16 + lx;  // query index i

  // Q as B-operand (K-dim = d): qf[g] covers d = g*32 .. g*32+31
  v16bf qf[2];
#pragma unroll
  for (int g = 0; g < 2; ++g)
#pragma unroll
    for (int e = 0; e < 16; ++e)
      qf[g][e] = qh[(size_t)(g * 32 + hi * 8 + (e & 7) + (e >> 3) * 16) * SEQ + col];

  v8f acc[4] = {{}, {}, {}, {}};
  float mrun = -1e30f, lrun = 0.0f;

  // ---- stage j-tile pair 0 into buffer 0 ----
#if HAVE_TDM
  if (wave == 0) {
    tdm_load_tile(kh, (unsigned)(size_t)(void*)sk[0], 1, 32, 64, SEQ, 64, SEQ);
    tdm_load_tile(vh, (unsigned)(size_t)(void*)sv[0], 1, 32, 64, SEQ, 64, SEQ);
    __builtin_amdgcn_s_wait_tensorcnt(0);
  }
#else
  {
    int row = threadIdx.x >> 2, c = (threadIdx.x & 3) * 8;
    *(uint4*)(&sk[0][row * 32 + c]) = *(const uint4*)(kh + (size_t)row * SEQ + c);
    *(uint4*)(&sv[0][row * 32 + c]) = *(const uint4*)(vh + (size_t)row * SEQ + c);
  }
#endif
  __syncthreads();

  int cur = 0;
  for (int jt = 0; jt < 128; jt += 2) {  // two 16-wide j tiles per step
    const int nxt = cur ^ 1;
    // ---- prefetch next j-tile pair into the other buffer ----
    if (jt + 2 < 128) {
      const int jn = (jt + 2) * 16;
#if HAVE_TDM
      if (wave == 0) {
        tdm_load_tile(kh + jn, (unsigned)(size_t)(void*)sk[nxt], 1, 32, 64, SEQ,
                      64, SEQ);
        tdm_load_tile(vh + jn, (unsigned)(size_t)(void*)sv[nxt], 1, 32, 64, SEQ,
                      64, SEQ);
      }
#else
      int row = threadIdx.x >> 2, c = (threadIdx.x & 3) * 8;
      *(uint4*)(&sk[nxt][row * 32 + c]) =
          *(const uint4*)(kh + (size_t)row * SEQ + jn + c);
      *(uint4*)(&sv[nxt][row * 32 + c]) =
          *(const uint4*)(vh + (size_t)row * SEQ + jn + c);
#endif
    }

    const __bf16* kb = sk[cur];
    const __bf16* vb = sv[cur];

    // S^T tiles from LDS: A = K^T (rows j, cols d), B = Q
    v8f s2[2] = {{}, {}};
#pragma unroll
    for (int t = 0; t < 2; ++t) {
#pragma unroll
      for (int g = 0; g < 2; ++g) {
        v16bf ka;
#pragma unroll
        for (int e = 0; e < 8; ++e) {
          ka[e]     = kb[(g * 32 + hi * 8 + e) * 32 + t * 16 + lx];
          ka[e + 8] = kb[(g * 32 + 16 + hi * 8 + e) * 32 + t * 16 + lx];
        }
        s2[t] = wmma_bf16(ka, qf[g], s2[t]);
      }
    }

    // Online softmax over j for fixed i = lx; lane^16 holds the other 16 j's.
    float p0[8], p1[8];
    float tmax = -1e30f;
#pragma unroll
    for (int r = 0; r < 8; ++r) {
      p0[r] = s2[0][r] * ATTN_SCALE;
      p1[r] = s2[1][r] * ATTN_SCALE;
      tmax = fmaxf(tmax, fmaxf(p0[r], p1[r]));
    }
    tmax = fmaxf(tmax, __shfl_xor(tmax, 16, 32));
    float mnew = fmaxf(mrun, tmax);
    float alpha = __expf(mrun - mnew);
    float rsum = 0.0f;
#pragma unroll
    for (int r = 0; r < 8; ++r) {
      p0[r] = __expf(p0[r] - mnew);
      p1[r] = __expf(p1[r] - mnew);
      rsum += p0[r] + p1[r];
    }
    rsum += __shfl_xor(rsum, 16, 32);
    lrun = lrun * alpha + rsum;
    mrun = mnew;

    // Pack P^T (already j-major) into the 32x16 B operand.
    v16bf pb;
#pragma unroll
    for (int r = 0; r < 8; ++r) {
      float o0 = __shfl_xor(p0[r], 16, 32);
      float o1 = __shfl_xor(p1[r], 16, 32);
      pb[r]     = (__bf16)(hi ? o1 : p0[r]);
      pb[r + 8] = (__bf16)(hi ? p1[r] : o0);
    }

    // O += V * P from LDS (A = V rows d, K-dim = j; contiguous reads)
#pragma unroll
    for (int blk = 0; blk < 4; ++blk) {
#pragma unroll
      for (int r = 0; r < 8; ++r) acc[blk][r] *= alpha;
      const __bf16* vrow = vb + (blk * 16 + lx) * 32;
      v16bf va;
#pragma unroll
      for (int e = 0; e < 8; ++e) {
        va[e]     = vrow[hi * 16 + e];
        va[e + 8] = vrow[hi * 16 + 8 + e];
      }
      acc[blk] = wmma_bf16(va, pb, acc[blk]);
    }

#if HAVE_TDM
    if (wave == 0) __builtin_amdgcn_s_wait_tensorcnt(0);
#endif
    __syncthreads();
    cur = nxt;
  }

  float inv = 1.0f / lrun;
  __bf16* aob = ao + (size_t)bidx * HIDDEN * SEQ;
#pragma unroll
  for (int blk = 0; blk < 4; ++blk)
#pragma unroll
    for (int r = 0; r < 8; ++r)
      aob[(size_t)(h * DHEAD + blk * 16 + hi * 8 + r) * SEQ + col] =
          (__bf16)(acc[blk][r] * inv);
}

// ---------------------------------------------------------------------------
// Kernel 3: out[b,o,n] = sum_c w_out[o,c] * ao[b,c,n] + b_out[o]   (f32 out)
// Block owns one 16-wide n-slice; ao panel [512 x 16] bf16 staged via TDM.
// ---------------------------------------------------------------------------
__global__ __launch_bounds__(256) void k_out_proj(const __bf16* __restrict__ ao,
                                                  const float* __restrict__ w,
                                                  const float* __restrict__ bias,
                                                  float* __restrict__ out) {
  __shared__ __attribute__((aligned(16))) __bf16 aos[HIDDEN * 16];  // 16 KB

  const int lane = threadIdx.x & 31;
  const int wave = threadIdx.x >> 5;
  const int hi = lane >> 4;
  const int lx = lane & 15;

  const int bidx = blockIdx.x >> 7;
  const int tn = blockIdx.x & 127;
  const __bf16* ag = ao + (size_t)bidx * HIDDEN * SEQ + tn * 16;

#if HAVE_TDM
  if (wave == 0) {
    tdm_load_tile(ag, (unsigned)(size_t)(void*)aos, 1, 16, HIDDEN, SEQ, HIDDEN,
                  SEQ);
    __builtin_amdgcn_s_wait_tensorcnt(0);
  }
#else
  {
#pragma unroll
    for (int rr = 0; rr < 2; ++rr) {
      int row = threadIdx.x * 2 + rr;
      *(uint4*)&aos[row * 16] = *(const uint4*)(ag + (size_t)row * SEQ);
      *(uint4*)&aos[row * 16 + 8] = *(const uint4*)(ag + (size_t)row * SEQ + 8);
    }
  }
#endif
  __syncthreads();

  const int tm0 = wave * 2;  // 16 row-tiles / 8 waves
  v8f acc[2] = {};

  for (int k0 = 0; k0 < HIDDEN; k0 += 32) {
    v16bf bf;
#pragma unroll
    for (int e = 0; e < 16; ++e)
      bf[e] = aos[(k0 + hi * 8 + (e & 7) + (e >> 3) * 16) * 16 + lx];
#pragma unroll
    for (int t = 0; t < 2; ++t) {
      const float* wrow = w + (size_t)((tm0 + t) * 16 + lx) * HIDDEN + k0;
      v16bf af;
#pragma unroll
      for (int e = 0; e < 8; ++e) {
        af[e]     = (__bf16)wrow[hi * 8 + e];
        af[e + 8] = (__bf16)wrow[16 + hi * 8 + e];
      }
      acc[t] = wmma_bf16(af, bf, acc[t]);
    }
  }

  float* ob = out + (size_t)bidx * DIM * SEQ + tn * 16 + lx;
#pragma unroll
  for (int t = 0; t < 2; ++t)
#pragma unroll
    for (int r = 0; r < 8; ++r) {
      int o = (tm0 + t) * 16 + hi * 8 + r;
      ob[(size_t)o * SEQ] = acc[t][r] + bias[o];
    }
}

// ---------------------------------------------------------------------------
extern "C" void kernel_launch(void* const* d_in, const int* in_sizes, int n_in,
                              void* d_out, int out_size, void* d_ws, size_t ws_size,
                              hipStream_t stream) {
  (void)in_sizes; (void)n_in; (void)out_size; (void)ws_size;
  const float* x     = (const float*)d_in[0];
  const float* w_qkv = (const float*)d_in[1];
  const float* w_out = (const float*)d_in[2];
  const float* b_out = (const float*)d_in[3];
  float* out = (float*)d_out;

  // ws: qkv bf16 [4][1536][2048] (25.2 MB) then attn-out bf16 [4][512][2048] (8.4 MB)
  __bf16* qkv = (__bf16*)d_ws;
  __bf16* ao  = qkv + (size_t)BATCH * QKV_ROWS * SEQ;

  k_qkv_proj<<<BATCH * (SEQ / 16), 256, 0, stream>>>(x, w_qkv, qkv);
  k_attention<<<(BATCH * HEADS * (SEQ / 16)) / 8, 256, 0, stream>>>(qkv, ao);
  k_out_proj<<<BATCH * (SEQ / 16), 256, 0, stream>>>(ao, w_out, b_out, out);
}
